// MultiHeadAttentionLayer_44478681317976
// MI455X (gfx1250) — matrix operand order repaired and checked by reading, not compiled, not run on hardware
//
#include <hip/hip_runtime.h>

typedef __attribute__((ext_vector_type(16))) __bf16 v16bf;
typedef __attribute__((ext_vector_type(8)))  __bf16 v8bf;
typedef __attribute__((ext_vector_type(8)))  float  v8f;

#define DEV __device__ __forceinline__

static constexpr int BB = 4, SS = 2048, DD = 512, HH = 16, DHH = 32;

DEV unsigned short f2bf_bits(float f) {
  union { float f; unsigned u; } cv{f};
  unsigned u = cv.u;
  unsigned r = u + 0x7FFFu + ((u >> 16) & 1u);   // round-to-nearest-even
  return (unsigned short)(r >> 16);
}
DEV __bf16 f2bf(float f) {
  union { unsigned short u; __bf16 b; } cv;
  cv.u = f2bf_bits(f);
  return cv.b;
}
// Build a v16bf A/B operand from two 8-element (16B) chunks.
DEV v16bf ld8x2(const unsigned short* p0, const unsigned short* p1) {
  v8bf lo = *(const v8bf*)p0;
  v8bf hi = *(const v8bf*)p1;
  v16bf r;
#pragma unroll
  for (int i = 0; i < 8; ++i) { r[i] = lo[i]; r[i + 8] = hi[i]; }
  return r;
}
DEV v8f wmma_bf16(v16bf a, v16bf b, v8f c) {
  return __builtin_amdgcn_wmma_f32_16x16x32_bf16(false, a, false, b, (short)0, c,
                                                 false, false);
}

// ---------------------------------------------------------------- conversions
__global__ void cvt_bf16_kernel(const float* __restrict__ src,
                                unsigned short* __restrict__ dst, int n) {
  int i = blockIdx.x * blockDim.x + threadIdx.x;
  int stride = gridDim.x * blockDim.x;
  for (; i < n; i += stride) dst[i] = f2bf_bits(src[i]);
}

// Wq/Wk/Wv [H][D][DH] f32  ->  wT [3][H][DH][D] bf16 (B-operand K runs contiguous)
__global__ void transpose_w_kernel(const float* __restrict__ wq,
                                   const float* __restrict__ wk,
                                   const float* __restrict__ wv,
                                   unsigned short* __restrict__ wT) {
  int z = blockIdx.z;
  const float* src = (z == 0) ? wq : (z == 1) ? wk : wv;
  unsigned short* dst = wT + (size_t)z * HH * DHH * DD;
  int i = blockIdx.x * blockDim.x + threadIdx.x;   // i = ((h*32 + k)*512 + d)
  if (i < HH * DHH * DD) {
    int d = i & (DD - 1);
    int k = (i >> 9) & (DHH - 1);
    int h = i >> 14;
    dst[i] = f2bf_bits(src[((size_t)h * DD + d) * DHH + k]);
  }
}

// ---------------------------------------------------------------- QKV projection
// grid (256 Mtiles of 32 rows, 16 heads, 3 mats), 1 wave/block.
// 32x32 register tile: 4 WMMAs per 8 b128 loads per K step.
__global__ __launch_bounds__(32)
void qkv_proj_kernel(const unsigned short* __restrict__ xbf,
                     const unsigned short* __restrict__ wT,
                     const float* __restrict__ bq, const float* __restrict__ bk,
                     const float* __restrict__ bv,
                     unsigned short* __restrict__ qbf,
                     unsigned short* __restrict__ kbf,
                     unsigned short* __restrict__ vt) {
  __shared__ __align__(16) float ldsf[16 * 33];
  const int mt = blockIdx.x, h = blockIdx.y, z = blockIdx.z;
  const int lane = threadIdx.x, half = lane >> 4, n = lane & 15;
  const int row0 = mt * 32;
  const unsigned short* w = wT + ((size_t)z * HH + h) * DHH * DD;
  const float* bias = (z == 0) ? bq : (z == 1) ? bk : bv;

  v8f c00 = {}, c01 = {}, c10 = {}, c11 = {};
  const unsigned short* ar0 = xbf + (size_t)(row0 + n) * DD + half * 8;
  const unsigned short* ar1 = ar0 + (size_t)16 * DD;
#pragma unroll 4
  for (int kk = 0; kk < DD; kk += 32) {
    v16bf a0 = ld8x2(ar0 + kk, ar0 + kk + 16);
    v16bf a1 = ld8x2(ar1 + kk, ar1 + kk + 16);
    const unsigned short* b1p = w + (size_t)n * DD + kk + half * 8;
    const unsigned short* b2p = b1p + 16 * DD;
    v16bf b1 = ld8x2(b1p, b1p + 16);
    v16bf b2 = ld8x2(b2p, b2p + 16);
    c00 = wmma_bf16(a0, b1, c00);
    c01 = wmma_bf16(a0, b2, c01);
    c10 = wmma_bf16(a1, b1, c10);
    c11 = wmma_bf16(a1, b2, c11);
  }
  const float bb1 = bias[h * DHH + n], bb2 = bias[h * DHH + 16 + n];
#pragma unroll
  for (int r = 0; r < 8; ++r) {
    c00[r] += bb1; c10[r] += bb1;
    c01[r] += bb2; c11[r] += bb2;
  }

  const int b = row0 / SS, s0 = row0 % SS;
  const int bh = b * HH + h;
  if (z < 2) {
    unsigned short* outp = (z == 0) ? qbf : kbf;
#pragma unroll
    for (int sub = 0; sub < 2; ++sub) {
      v8f cc1 = sub ? c10 : c00;
      v8f cc2 = sub ? c11 : c01;
      if (sub) __syncthreads();
#pragma unroll
      for (int r = 0; r < 8; ++r) {
        int m = r + half * 8;
        ldsf[m * 33 + n]      = cc1[r];
        ldsf[m * 33 + 16 + n] = cc2[r];
      }
      __syncthreads();
      unsigned short* dst =
          outp + ((size_t)bh * SS + s0 + sub * 16 + n) * DHH + half * 16;
      v8bf v0, v1;
#pragma unroll
      for (int j = 0; j < 8; ++j) {
        v0[j] = f2bf(ldsf[n * 33 + half * 16 + j]);
        v1[j] = f2bf(ldsf[n * 33 + half * 16 + 8 + j]);
      }
      *(v8bf*)dst       = v0;
      *(v8bf*)(dst + 8) = v1;
    }
  } else {
    // V stored transposed [bh][dh][s]: C-tile columns are contiguous V^T rows.
    v8bf v0, v1, v2, v3;
#pragma unroll
    for (int j = 0; j < 8; ++j) {
      v0[j] = f2bf(c00[j]); v1[j] = f2bf(c01[j]);
      v2[j] = f2bf(c10[j]); v3[j] = f2bf(c11[j]);
    }
    unsigned short* d1 = vt + ((size_t)bh * DHH + n) * SS + s0 + half * 8;
    unsigned short* d2 = vt + ((size_t)bh * DHH + 16 + n) * SS + s0 + half * 8;
    *(v8bf*)d1        = v0;
    *(v8bf*)d2        = v1;
    *(v8bf*)(d1 + 16) = v2;      // rows s0+16..s0+31
    *(v8bf*)(d2 + 16) = v3;
  }
}

// ---------------------------------------------------------------- flash attention
// grid (B*H*S/16), 1 wave/block. Online softmax over 2048 keys, 32 per step.
__global__ __launch_bounds__(32)
void flash_attn_kernel(const unsigned short* __restrict__ qbf,
                       const unsigned short* __restrict__ kbf,
                       const unsigned short* __restrict__ vt,
                       unsigned short* __restrict__ mh) {
  __shared__ __align__(16) float ldsf[16 * 33];
  unsigned short* ldsp = (unsigned short*)ldsf;   // 16x32 bf16 P staging

  const int blk = blockIdx.x;
  const int bh = blk >> 7;               // /(S/16)
  const int s0 = (blk & 127) << 4;
  const int b = bh >> 4, h = bh & 15;
  const int lane = threadIdx.x, half = lane >> 4, n = lane & 15;

  const unsigned short* qrow = qbf + ((size_t)bh * SS + s0 + n) * DHH + half * 16;
  const v16bf qa = ld8x2(qrow, qrow + 8);         // K-dim == DH == 32: one WMMA

  const unsigned short* krow = kbf + ((size_t)bh * SS + n) * DHH + half * 16;
  const unsigned short* vrow1 = vt + ((size_t)bh * DHH + n) * SS + half * 16;
  const unsigned short* vrow2 = vt + ((size_t)bh * DHH + 16 + n) * SS + half * 16;

  float rmax[8], rsum[8];
  v8f o1 = {}, o2 = {};
#pragma unroll
  for (int r = 0; r < 8; ++r) { rmax[r] = -1e30f; rsum[r] = 0.f; }
  const float scale = 0.17677669529663687f;       // 1/sqrt(32)

  for (int t0 = 0; t0 < SS; t0 += 32) {
    // ---- scores: two 16x16 tiles
    const unsigned short* kr1 = krow + (size_t)t0 * DHH;
    const unsigned short* kr2 = kr1 + 16 * DHH;
    v16bf kb1 = ld8x2(kr1, kr1 + 8);
    v16bf kb2 = ld8x2(kr2, kr2 + 8);
    v8f z = {};
    v8f s1 = wmma_bf16(qa, kb1, z);
    v8f s2 = wmma_bf16(qa, kb2, z);

    // prefetch next key/value block into cache (global_prefetch_b8)
    if (t0 + 32 < SS) {
      __builtin_prefetch(kr1 + 32 * DHH, 0, 3);
      __builtin_prefetch(vrow1 + t0 + 32, 0, 3);
      __builtin_prefetch(vrow2 + t0 + 32, 0, 3);
    }

    // ---- online softmax (row = vgpr within 16-lane half)
    float nm[8];
#pragma unroll
    for (int r = 0; r < 8; ++r) {
      s1[r] *= scale; s2[r] *= scale;
      nm[r] = fmaxf(s1[r], s2[r]);
    }
#pragma unroll
    for (int off = 8; off >= 1; off >>= 1)
#pragma unroll
      for (int r = 0; r < 8; ++r) nm[r] = fmaxf(nm[r], __shfl_xor(nm[r], off, 32));

    float p1[8], p2[8], ls[8];
#pragma unroll
    for (int r = 0; r < 8; ++r) {
      float m2 = fmaxf(rmax[r], nm[r]);
      float corr = __expf(rmax[r] - m2);
      rmax[r] = m2;
      p1[r] = __expf(s1[r] - m2);
      p2[r] = __expf(s2[r] - m2);
      ls[r] = p1[r] + p2[r];
      o1[r] *= corr; o2[r] *= corr;
      rsum[r] *= corr;
    }
#pragma unroll
    for (int off = 8; off >= 1; off >>= 1)
#pragma unroll
      for (int r = 0; r < 8; ++r) ls[r] += __shfl_xor(ls[r], off, 32);
#pragma unroll
    for (int r = 0; r < 8; ++r) rsum[r] += ls[r];

    // ---- repack P (col-layout) -> bf16 A-layout via LDS
    __syncthreads();                      // prior A-reads done before overwrite
#pragma unroll
    for (int r = 0; r < 8; ++r) {
      int m = r + half * 8;
      ldsp[m * 32 + n]      = f2bf_bits(p1[r]);
      ldsp[m * 32 + 16 + n] = f2bf_bits(p2[r]);
    }
    __syncthreads();
    v16bf pa = ld8x2(ldsp + n * 32 + half * 16, ldsp + n * 32 + half * 16 + 8);

    // ---- O += P * V  (V^T rows give contiguous K runs)
    const unsigned short* vr1 = vrow1 + t0;
    const unsigned short* vr2 = vrow2 + t0;
    v16bf vb1 = ld8x2(vr1, vr1 + 8);
    v16bf vb2 = ld8x2(vr2, vr2 + 8);
    o1 = wmma_bf16(pa, vb1, o1);
    o2 = wmma_bf16(pa, vb2, o2);
  }

  // ---- normalize, transpose via LDS, store bf16 multihead [B,S,512] @ col h*32
#pragma unroll
  for (int r = 0; r < 8; ++r) {
    float inv = 1.0f / rsum[r];
    o1[r] *= inv; o2[r] *= inv;
  }
  __syncthreads();
#pragma unroll
  for (int r = 0; r < 8; ++r) {
    int m = r + half * 8;
    ldsf[m * 33 + n]      = o1[r];
    ldsf[m * 33 + 16 + n] = o2[r];
  }
  __syncthreads();
  unsigned short* dst =
      mh + ((size_t)(b * SS + s0 + n)) * DD + h * DHH + half * 16;
  v8bf v0, v1;
#pragma unroll
  for (int j = 0; j < 8; ++j) {
    v0[j] = f2bf(ldsf[n * 33 + half * 16 + j]);
    v1[j] = f2bf(ldsf[n * 33 + half * 16 + 8 + j]);
  }
  *(v8bf*)dst       = v0;
  *(v8bf*)(dst + 8) = v1;
}

// ---------------------------------------------------------------- output proj
// y = mh @ Wo^T + bo : M=8192, N=512, K=512. 32x32 register tile per wave.
__global__ __launch_bounds__(32)
void out_proj_kernel(const unsigned short* __restrict__ mh,
                     const unsigned short* __restrict__ wo,
                     const float* __restrict__ bo, float* __restrict__ out) {
  __shared__ __align__(16) float ldsf[16 * 33];
  const int mt = blockIdx.x, nt = blockIdx.y;
  const int lane = threadIdx.x, half = lane >> 4, n = lane & 15;
  const int row0 = mt * 32, n0 = nt * 32;

  v8f c00 = {}, c01 = {}, c10 = {}, c11 = {};
  const unsigned short* ar0 = mh + (size_t)(row0 + n) * DD + half * 8;
  const unsigned short* ar1 = ar0 + (size_t)16 * DD;
#pragma unroll 4
  for (int kk = 0; kk < DD; kk += 32) {
    v16bf a0 = ld8x2(ar0 + kk, ar0 + kk + 16);
    v16bf a1 = ld8x2(ar1 + kk, ar1 + kk + 16);
    const unsigned short* b1p = wo + (size_t)(n0 + n) * DD + kk + half * 8;
    const unsigned short* b2p = b1p + 16 * DD;
    v16bf b1 = ld8x2(b1p, b1p + 16);
    v16bf b2 = ld8x2(b2p, b2p + 16);
    c00 = wmma_bf16(a0, b1, c00);
    c01 = wmma_bf16(a0, b2, c01);
    c10 = wmma_bf16(a1, b1, c10);
    c11 = wmma_bf16(a1, b2, c11);
  }
  const float bb1 = bo[n0 + n], bb2 = bo[n0 + 16 + n];
#pragma unroll
  for (int sub = 0; sub < 2; ++sub) {
    v8f cc1 = sub ? c10 : c00;
    v8f cc2 = sub ? c11 : c01;
    if (sub) __syncthreads();
#pragma unroll
    for (int r = 0; r < 8; ++r) {
      int m = r + half * 8;
      ldsf[m * 33 + n]      = cc1[r] + bb1;
      ldsf[m * 33 + 16 + n] = cc2[r] + bb2;
    }
    __syncthreads();
    float* dst = out + (size_t)(row0 + sub * 16 + n) * DD + n0 + half * 16;
#pragma unroll
    for (int j = 0; j < 4; ++j) {
      float4 t;
      t.x = ldsf[n * 33 + half * 16 + 4 * j + 0];
      t.y = ldsf[n * 33 + half * 16 + 4 * j + 1];
      t.z = ldsf[n * 33 + half * 16 + 4 * j + 2];
      t.w = ldsf[n * 33 + half * 16 + 4 * j + 3];
      ((float4*)dst)[j] = t;
    }
  }
}

// ---------------------------------------------------------------- launch
extern "C" void kernel_launch(void* const* d_in, const int* in_sizes, int n_in,
                              void* d_out, int out_size, void* d_ws, size_t ws_size,
                              hipStream_t stream) {
  (void)in_sizes; (void)n_in; (void)out_size; (void)ws_size;
  const float* x  = (const float*)d_in[0];
  const float* Wq = (const float*)d_in[1];
  const float* bq = (const float*)d_in[2];
  const float* Wk = (const float*)d_in[3];
  const float* bk = (const float*)d_in[4];
  const float* Wv = (const float*)d_in[5];
  const float* bv = (const float*)d_in[6];
  const float* Wo = (const float*)d_in[7];
  const float* bo = (const float*)d_in[8];
  float* out = (float*)d_out;

  const size_t nX = (size_t)BB * SS * DD;        // 4,194,304
  const size_t nW = (size_t)HH * DHH * DD;       // 262,144 per matrix
  const size_t nQ = (size_t)BB * HH * SS * DHH;  // 4,194,304

  unsigned char* ws = (unsigned char*)d_ws;
  size_t off = 0;
  auto carve = [&](size_t bytes) {
    unsigned char* p = ws + off;
    off = (off + bytes + 255) & ~(size_t)255;
    return p;
  };
  unsigned short* xbf  = (unsigned short*)carve(nX * 2);
  unsigned short* wT   = (unsigned short*)carve(3 * nW * 2);
  unsigned short* wobf = (unsigned short*)carve((size_t)DD * DD * 2);
  unsigned short* qbf  = (unsigned short*)carve(nQ * 2);
  unsigned short* kbf  = (unsigned short*)carve(nQ * 2);
  unsigned short* vt   = (unsigned short*)carve(nQ * 2);
  unsigned short* mh   = (unsigned short*)carve(nX * 2);

  cvt_bf16_kernel<<<4096, 256, 0, stream>>>(x, xbf, (int)nX);
  cvt_bf16_kernel<<<1024, 256, 0, stream>>>(Wo, wobf, DD * DD);
  transpose_w_kernel<<<dim3(1024, 1, 3), 256, 0, stream>>>(Wq, Wk, Wv, wT);

  qkv_proj_kernel<<<dim3(BB * SS / 32, HH, 3), 32, 0, stream>>>(
      xbf, wT, bq, bk, bv, qbf, kbf, vt);

  flash_attn_kernel<<<BB * HH * (SS / 16), 32, 0, stream>>>(qbf, kbf, vt, mh);

  out_proj_kernel<<<dim3(BB * SS / 32, DD / 32), 32, 0, stream>>>(mh, wobf, bo,
                                                                  out);
}
